// Experts_19430432047439
// MI455X (gfx1250) — compile-verified, matrix operand order checked by reading, not run on hardware
//
#include <hip/hip_runtime.h>
#include <hip/hip_bf16.h>
#include <stdint.h>

#define N_NODES   65536
#define N_EDGES   1048576
#define HID       128
#define NUM_FEAT  128
#define NUM_LAYERS 3
#define NUM_EXPERTS 4
#define NUM_GRAPHS 512
#define NUM_CLASSES 10

typedef __attribute__((ext_vector_type(16))) __bf16 v16bf;
typedef __attribute__((ext_vector_type(8)))  float  v8f;
typedef __attribute__((ext_vector_type(4)))  unsigned int v4u;
typedef __attribute__((ext_vector_type(8)))  int  v8i;
typedef __attribute__((ext_vector_type(4)))  int  v4i;

#if defined(__has_builtin)
#  if __has_builtin(__builtin_amdgcn_tensor_load_to_lds)
#    define HAVE_TDM 1
#  else
#    define HAVE_TDM 0
#  endif
#else
#  define HAVE_TDM 0
#endif

// ---------- helpers ----------
static __device__ __forceinline__ unsigned short f2bf(float f) {
  unsigned int u = __float_as_uint(f);
  unsigned int r = (u + 0x7FFFu + ((u >> 16) & 1u)) >> 16;
  return (unsigned short)r;
}
static __device__ __forceinline__ float bf2f(unsigned short h) {
  return __uint_as_float(((unsigned int)h) << 16);
}
static __device__ __forceinline__ unsigned int hash2(unsigned int a, unsigned int b) {
  unsigned int h = a * 0x9E3779B9u + b * 0x85EBCA6Bu + 0x7F4A7C15u;
  h ^= h >> 16; h *= 0x7FEB352Du;
  h ^= h >> 15; h *= 0x846CA68Bu;
  h ^= h >> 16;
  return h;
}
// Gumbel noise g; hard-concrete forward value is 1[(logit+g) > 0]
static __device__ __forceinline__ float gumbel_for(unsigned int seed, unsigned int idx) {
  unsigned int h = hash2(seed, idx);
  float u = (float)(h >> 8) * (1.0f / 16777216.0f);
  u = fminf(fmaxf(u, 1e-6f), 1.0f - 1e-6f);
  return -__logf(-__logf(u + 1e-20f) + 1e-20f);
}

#if HAVE_TDM
// Issue one TDM DMA: contiguous-row 2D tile (rows x K bf16) from global -> LDS.
// D# layout per CDNA5 ISA.
static __device__ __forceinline__ void tdm_load_tile(unsigned lds_off,
                                                     const void* gaddr,
                                                     int K, int rows) {
  unsigned long long ga = (unsigned long long)gaddr;
  v4u g0;
  g0.x = 1u;                                   // count=1, user descriptor
  g0.y = lds_off;                              // lds_addr (bytes)
  g0.z = (unsigned)ga;                         // global_addr[31:0]
  g0.w = (unsigned)((ga >> 32) & 0x01FFFFFFu) | 0x80000000u;  // addr[56:32] | type=2
  v8i g1;
  g1[0] = (int)(1u << 16);                     // workgroup_mask=0, data_size=1 (2B)
  g1[1] = (int)(((unsigned)K & 0xFFFFu) << 16);            // tensor_dim0[15:0]
  g1[2] = (int)((((unsigned)K >> 16) & 0xFFFFu) |          // tensor_dim0[31:16]
                (((unsigned)rows & 0xFFFFu) << 16));       // tensor_dim1[15:0]
  g1[3] = (int)((((unsigned)rows >> 16) & 0xFFFFu) |       // tensor_dim1[31:16]
                (((unsigned)K & 0xFFFFu) << 16));          // tile_dim0
  g1[4] = (int)((unsigned)rows & 0xFFFFu);                 // tile_dim1 (tile_dim2=0)
  g1[5] = K;                                               // tensor_dim0_stride[31:0]
  g1[6] = 0;
  g1[7] = 0;
  v4i z4 = {0, 0, 0, 0};
#if __clang_major__ >= 23
  v8i z8 = {0, 0, 0, 0, 0, 0, 0, 0};
  __builtin_amdgcn_tensor_load_to_lds(g0, g1, z4, z4, z8, 0);
#else
  __builtin_amdgcn_tensor_load_to_lds(g0, g1, z4, z4, 0);
#endif
}
#endif

// ---------- elementwise / prep kernels ----------
__global__ void convert_f32_bf16(const float* __restrict__ in,
                                 unsigned short* __restrict__ out, long n) {
  long i = (long)blockIdx.x * blockDim.x + threadIdx.x;
  if (i < n) out[i] = f2bf(in[i]);
}

// W (K x N, f32, row-major) -> WT (N x K, bf16)
__global__ void wprep(const float* __restrict__ W, unsigned short* __restrict__ WT,
                      int K, int N) {
  long i = (long)blockIdx.x * blockDim.x + threadIdx.x;
  if (i >= (long)K * N) return;
  int kk = (int)(i / N), n = (int)(i - (long)kk * N);
  WT[(size_t)n * K + kk] = f2bf(W[i]);
}

// z = (1+eps[l]) * h + agg   (bf16 out)
__global__ void gin_combine(const unsigned short* __restrict__ hb,
                            const float* __restrict__ agg,
                            const float* __restrict__ eps, int l,
                            unsigned short* __restrict__ zb) {
  long i = (long)blockIdx.x * blockDim.x + threadIdx.x;
  if (i >= (long)N_NODES * HID) return;
  zb[i] = f2bf((1.0f + eps[l]) * bf2f(hb[i]) + agg[i]);
}

// agg[dst] += w * h[src]   (one edge per 32 threads, 4 feats/thread, bf16 reads)
__global__ void scatter_edges(const unsigned short* __restrict__ hb,
                              const int* __restrict__ src, const int* __restrict__ dst,
                              const float* __restrict__ ew, int ewStride,
                              float* __restrict__ agg) {
  long tid = (long)blockIdx.x * blockDim.x + threadIdx.x;
  long e = tid >> 5;
  if (e >= N_EDGES) return;
  int q = (int)(tid & 31);
  float w = 1.0f;
  if (ew) {
    w = ew[e * ewStride];
    if (w == 0.0f) return;   // binary mask: skip dead edges entirely
  }
  int s = src[e], d = dst[e];
  const unsigned int* p =
      reinterpret_cast<const unsigned int*>(hb + (size_t)s * HID + q * 4);
  unsigned int u0 = p[0], u1 = p[1];
  float* ag = agg + (size_t)d * HID + q * 4;
  atomicAdd(ag + 0, bf2f((unsigned short)(u0 & 0xFFFFu)) * w);
  atomicAdd(ag + 1, bf2f((unsigned short)(u0 >> 16)) * w);
  atomicAdd(ag + 2, bf2f((unsigned short)(u1 & 0xFFFFu)) * w);
  atomicAdd(ag + 3, bf2f((unsigned short)(u1 >> 16)) * w);
}

// edge_feat = [Z[src], Z[dst]]  (bf16, 1M x 256)
__global__ void build_edge_feat(const unsigned short* __restrict__ Zb,
                                const int* __restrict__ src, const int* __restrict__ dst,
                                unsigned short* __restrict__ ef) {
  long tid = (long)blockIdx.x * blockDim.x + threadIdx.x;
  long e = tid >> 5;
  if (e >= N_EDGES) return;
  int q = (int)(tid & 31);
  int node = (q < 16) ? src[e] : dst[e];
  int off = (q & 15) * 8;
  uint4 v = *reinterpret_cast<const uint4*>(Zb + (size_t)node * HID + off);
  *reinterpret_cast<uint4*>(ef + (size_t)e * 256 + q * 8) = v;
}

// masked_x = x * node_mask * feat_mask  (masks are binary f32 in d_out)
__global__ void masked_x_kernel(const unsigned short* __restrict__ xb,
                                const float* __restrict__ nodeMask,  // + k, stride 4
                                const float* __restrict__ featMask,  // + k*128, row stride 512
                                unsigned short* __restrict__ mxb) {
  long i = (long)blockIdx.x * blockDim.x + threadIdx.x;
  if (i >= (long)N_NODES * HID) return;
  long n = i >> 7;
  int c = (int)(i & 127);
  float v = bf2f(xb[i]) * nodeMask[n * 4] * featMask[n * 512 + c];
  mxb[i] = f2bf(v);
}

// ---- shared A-tile staging: one 16-row x K bf16 tile into LDS per block ----
static __device__ __forceinline__ void stage_A_tile(unsigned short* sA,
                                                    const unsigned short* Abase,
                                                    int K) {
#if HAVE_TDM
  if (threadIdx.x < 32) {   // one wave issues the DMA (EXEC ignored by TDM)
    tdm_load_tile((unsigned)(unsigned long long)(void*)sA, (const void*)Abase, K, 16);
    __builtin_amdgcn_s_wait_tensorcnt(0);
  }
  __syncthreads();
#else
  const int n16 = (16 * K) / 8;  // uint4 chunks (tile rows are contiguous)
  for (int i = threadIdx.x; i < n16; i += 256)
    reinterpret_cast<uint4*>(sA)[i] = reinterpret_cast<const uint4*>(Abase)[i];
  __syncthreads();
#endif
}

// ---------- WMMA GEMM: C = act(A @ W + b); block = 16 rows x 128 cols ----------
// A: MxK bf16 (staged to LDS), WT: NxK bf16 (N==128), 8 waves = 8 N-tiles.
__global__ void gemm_bf16_wmma(const unsigned short* __restrict__ A,
                               const unsigned short* __restrict__ WT,
                               const float* __restrict__ bias,
                               unsigned short* __restrict__ outB, int strideB,
                               float* __restrict__ outF, int strideF,
                               int M, int K, int N, int doRelu, int doHC,
                               unsigned int seed) {
  const int lane = threadIdx.x & 31;
  const int wid  = threadIdx.x >> 5;   // N-tile 0..7
  const int half = lane >> 4;
  const int mloc = lane & 15;
  const int tm = blockIdx.x;
  const int tn = wid;
  __shared__ unsigned short sA[16 * 128];

  stage_A_tile(sA, A + (size_t)tm * 16 * K, K);

  union V16 { unsigned int d[8]; v16bf bf; };
  V16 a, b;
  v8f acc = {0.f, 0.f, 0.f, 0.f, 0.f, 0.f, 0.f, 0.f};
  const unsigned short* Bp0 = WT + (size_t)(tn * 16 + mloc) * K;
  const unsigned short* As0 = sA + mloc * K;
  for (int kb = 0; kb < K; kb += 32) {
#pragma unroll
    for (int v = 0; v < 8; ++v) {
      // 16-bit A-matrix fragment layout: pairs of K per dword (ISA 7.12.2)
      int ko = ((v & 3) << 1) | ((v >> 2) << 4) | (half << 3);
      a.d[v] = *reinterpret_cast<const unsigned int*>(As0 + kb + ko);  // ds_load
      b.d[v] = *reinterpret_cast<const unsigned int*>(Bp0 + kb + ko);  // global
    }
    acc = __builtin_amdgcn_wmma_f32_16x16x32_bf16(false, a.bf, false, b.bf,
                                                  (short)0, acc, false, false);
  }
  const int c = tn * 16 + mloc;
  const float bv = bias ? bias[c] : 0.0f;
#pragma unroll
  for (int v = 0; v < 8; ++v) {
    int r = tm * 16 + v + (half << 3);
    float val = acc[v] + bv;
    if (doRelu) val = fmaxf(val, 0.0f);
    if (doHC) {
      float g = gumbel_for(seed, (unsigned int)(r * N + c));
      outF[(size_t)r * strideF + c] = ((val + g) > 0.0f) ? 1.0f : 0.0f;
    } else {
      if (outB) outB[(size_t)r * strideB + c] = f2bf(val);
      if (outF) outF[(size_t)r * strideF + c] = val;
    }
  }
}

// ---------- fused mask head: mask = HC(relu(A@W1+b1) @ w2 + b2) ----------
// block = 256 (8 waves): 16 rows x 128 hidden, A staged in LDS, no intermediate.
__global__ void maskhead_wmma(const unsigned short* __restrict__ A,   // M x K
                              const unsigned short* __restrict__ W1T, // 128 x K
                              const float* __restrict__ b1,           // 128
                              const float* __restrict__ w2,           // 128 (f32)
                              const float* __restrict__ b2,           // scalar
                              float* __restrict__ maskOut, int maskStride,
                              int K, unsigned int seed) {
  const int lane = threadIdx.x & 31;
  const int wid  = threadIdx.x >> 5;   // hidden-column tile 0..7
  const int half = lane >> 4;
  const int mloc = lane & 15;
  __shared__ unsigned short sA[16 * 256];
  __shared__ float lds[8][16];

  stage_A_tile(sA, A + (size_t)blockIdx.x * 16 * K, K);

  union V16 { unsigned int d[8]; v16bf bf; };
  V16 a, b;
  v8f acc = {0.f, 0.f, 0.f, 0.f, 0.f, 0.f, 0.f, 0.f};
  const unsigned short* Bp0 = W1T + (size_t)(wid * 16 + mloc) * K;
  const unsigned short* As0 = sA + mloc * K;
  for (int kb = 0; kb < K; kb += 32) {
#pragma unroll
    for (int v = 0; v < 8; ++v) {
      int ko = ((v & 3) << 1) | ((v >> 2) << 4) | (half << 3);
      a.d[v] = *reinterpret_cast<const unsigned int*>(As0 + kb + ko);
      b.d[v] = *reinterpret_cast<const unsigned int*>(Bp0 + kb + ko);
    }
    acc = __builtin_amdgcn_wmma_f32_16x16x32_bf16(false, a.bf, false, b.bf,
                                                  (short)0, acc, false, false);
  }
  const int c = wid * 16 + mloc;
  const float b1v = b1[c];
  const float w2v = w2[c];
#pragma unroll
  for (int v = 0; v < 8; ++v) {
    float s = fmaxf(acc[v] + b1v, 0.0f) * w2v;
    s += __shfl_xor(s, 1);
    s += __shfl_xor(s, 2);
    s += __shfl_xor(s, 4);
    s += __shfl_xor(s, 8);           // reduce 16 columns within half-wave
    if (mloc == 0) lds[wid][v + (half << 3)] = s;
  }
  __syncthreads();
  int t = threadIdx.x;
  if (t < 16) {
    float tot = 0.0f;
#pragma unroll
    for (int w = 0; w < 8; ++w) tot += lds[w][t];
    float val = tot + b2[0];
    unsigned int r = blockIdx.x * 16 + t;
    float g = gumbel_for(seed, r);
    maskOut[(size_t)r * maskStride] = ((val + g) > 0.0f) ? 1.0f : 0.0f;
  }
}

// ---------- pooling / classifier ----------
__global__ void pool_accum(const unsigned short* __restrict__ hb,
                           const int* __restrict__ batch,
                           float* __restrict__ sums, float* __restrict__ cnt) {
  long tid = (long)blockIdx.x * blockDim.x + threadIdx.x;
  long n = tid >> 5;
  if (n >= N_NODES) return;
  int q = (int)(tid & 31);
  int g = batch[n];
  const unsigned int* p =
      reinterpret_cast<const unsigned int*>(hb + (size_t)n * HID + q * 4);
  unsigned int u0 = p[0], u1 = p[1];
  float* sg = sums + (size_t)g * HID + q * 4;
  atomicAdd(sg + 0, bf2f((unsigned short)(u0 & 0xFFFFu)));
  atomicAdd(sg + 1, bf2f((unsigned short)(u0 >> 16)));
  atomicAdd(sg + 2, bf2f((unsigned short)(u1 & 0xFFFFu)));
  atomicAdd(sg + 3, bf2f((unsigned short)(u1 >> 16)));
  if (q == 0) atomicAdd(cnt + g, 1.0f);
}

__global__ void pool_final(const float* __restrict__ sums, const float* __restrict__ cnt,
                           float* __restrict__ out, int rowStride) {
  int tid = blockIdx.x * blockDim.x + threadIdx.x;
  if (tid >= NUM_GRAPHS * HID) return;
  int g = tid >> 7, c = tid & 127;
  out[(size_t)g * rowStride + c] = sums[tid] / fmaxf(cnt[g], 1.0f);
}

// logits = hs @ cls_w + cls_b   (one wave per graph)
__global__ void logits_kernel(const float* __restrict__ hs, int rowStride,
                              const float* __restrict__ W, const float* __restrict__ bvec,
                              float* __restrict__ out, int outStride) {
  int g = blockIdx.x;
  int lane = threadIdx.x & 31;
  float hv[4];
#pragma unroll
  for (int j = 0; j < 4; ++j) hv[j] = hs[(size_t)g * rowStride + lane * 4 + j];
  for (int c = 0; c < NUM_CLASSES; ++c) {
    float s = 0.0f;
#pragma unroll
    for (int j = 0; j < 4; ++j) s += hv[j] * W[(lane * 4 + j) * NUM_CLASSES + c];
    s += __shfl_xor(s, 1);
    s += __shfl_xor(s, 2);
    s += __shfl_xor(s, 4);
    s += __shfl_xor(s, 8);
    s += __shfl_xor(s, 16);
    if (lane == 0) out[(size_t)g * outStride + c] = s + bvec[c];
  }
}

// ---------- host orchestration ----------
static inline long cdiv(long a, long b) { return (a + b - 1) / b; }

// One 3-layer GIN encoder pass. h0b: input bf16. Final activations land in hbA.
static void run_encoder(const unsigned short* h0b, unsigned short* hbA, unsigned short* hbB,
                        unsigned short* zb, unsigned short* t1b, float* agg,
                        const unsigned short* W1T, const unsigned short* W2T,  // 3 x 128x128
                        const float* b1, const float* b2, const float* eps,
                        const int* src, const int* dst,
                        const float* emask, int emaskStride, hipStream_t stream) {
  const unsigned short* cur = h0b;
  unsigned short* outs[3] = {hbB, hbA, hbB};
  outs[0] = hbA; outs[1] = hbB; outs[2] = hbA;   // final = hbA
  for (int l = 0; l < NUM_LAYERS; ++l) {
    hipMemsetAsync(agg, 0, (size_t)N_NODES * HID * sizeof(float), stream);
    scatter_edges<<<(unsigned)cdiv((long)N_EDGES * 32, 256), 256, 0, stream>>>(
        cur, src, dst, emask, emaskStride, agg);
    gin_combine<<<(unsigned)cdiv((long)N_NODES * HID, 256), 256, 0, stream>>>(
        cur, agg, eps, l, zb);
    // t1 = relu(z @ W1 + b1)
    gemm_bf16_wmma<<<N_NODES / 16, 256, 0, stream>>>(
        zb, W1T + (size_t)l * HID * HID, b1 + (size_t)l * HID,
        t1b, HID, nullptr, 0, N_NODES, HID, HID, 1, 0, 0u);
    // h = act(t1 @ W2 + b2); relu except last layer
    gemm_bf16_wmma<<<N_NODES / 16, 256, 0, stream>>>(
        t1b, W2T + (size_t)l * HID * HID, b2 + (size_t)l * HID,
        outs[l], HID, nullptr, 0, N_NODES, HID, HID, (l < NUM_LAYERS - 1) ? 1 : 0, 0, 0u);
    cur = outs[l];
  }
}

extern "C" void kernel_launch(void* const* d_in, const int* in_sizes, int n_in,
                              void* d_out, int out_size, void* d_ws, size_t ws_size,
                              hipStream_t stream) {
  // ---- inputs (setup_inputs order) ----
  const float* x       = (const float*)d_in[0];
  const int*   eidx    = (const int*)d_in[1];
  const int*   src     = eidx;
  const int*   dst     = eidx + N_EDGES;
  const int*   batch   = (const int*)d_in[2];
  const float* cw1 = (const float*)d_in[3];  const float* cb1 = (const float*)d_in[4];
  const float* cw2 = (const float*)d_in[5];  const float* cb2 = (const float*)d_in[6];
  const float* ceps = (const float*)d_in[7];
  const float* kw1 = (const float*)d_in[8];  const float* kb1 = (const float*)d_in[9];
  const float* kw2 = (const float*)d_in[10]; const float* kb2 = (const float*)d_in[11];
  const float* keps = (const float*)d_in[12];
  const float* nm_w1 = (const float*)d_in[13]; const float* nm_b1 = (const float*)d_in[14];
  const float* nm_w2 = (const float*)d_in[15]; const float* nm_b2 = (const float*)d_in[16];
  const float* em_w1 = (const float*)d_in[17]; const float* em_b1 = (const float*)d_in[18];
  const float* em_w2 = (const float*)d_in[19]; const float* em_b2 = (const float*)d_in[20];
  const float* fm_w1 = (const float*)d_in[21]; const float* fm_b1 = (const float*)d_in[22];
  const float* fm_w2 = (const float*)d_in[23]; const float* fm_b2 = (const float*)d_in[24];
  const float* cls_w = (const float*)d_in[25]; const float* cls_b = (const float*)d_in[26];

  // ---- outputs (flat, concatenated in return order) ----
  float* out = (float*)d_out;
  float* out_logits   = out;                         // (512, 4, 10)
  float* out_hstable  = out + 512 * 4 * 10;          // (512, 4, 128)
  float* out_horig    = out_hstable + 512 * 4 * 128; // (512, 128)
  float* out_nodemask = out_horig + 512 * 128;       // (65536, 4, 1)
  float* out_edgemask = out_nodemask + (size_t)N_NODES * 4;   // (1M, 4, 1)
  float* out_featmask = out_edgemask + (size_t)N_EDGES * 4;   // (65536, 4, 128)

  // ---- workspace carve-up ----
  char* wp = (char*)d_ws;
  auto alloc = [&](size_t bytes) -> char* {
    char* p = wp;
    wp += (bytes + 255) & ~(size_t)255;
    return p;
  };
  const size_t ACT = (size_t)N_NODES * HID;
  unsigned short* xb   = (unsigned short*)alloc(ACT * 2);
  unsigned short* Zb   = (unsigned short*)alloc(ACT * 2);
  unsigned short* hbA  = (unsigned short*)alloc(ACT * 2);
  unsigned short* hbB  = (unsigned short*)alloc(ACT * 2);
  unsigned short* zb   = (unsigned short*)alloc(ACT * 2);
  unsigned short* t1b  = (unsigned short*)alloc(ACT * 2);
  unsigned short* mxb  = (unsigned short*)alloc(ACT * 2);
  float*          agg  = (float*)alloc(ACT * 4);
  unsigned short* efb  = (unsigned short*)alloc((size_t)N_EDGES * 256 * 2);
  float* pool = (float*)alloc((size_t)NUM_GRAPHS * HID * 4);
  float* cnt  = (float*)alloc((size_t)NUM_GRAPHS * 4);
  const size_t WM = (size_t)HID * HID;  // 16384
  unsigned short* cw1T = (unsigned short*)alloc(3 * WM * 2);
  unsigned short* cw2T = (unsigned short*)alloc(3 * WM * 2);
  unsigned short* kw1T = (unsigned short*)alloc(3 * WM * 2);
  unsigned short* kw2T = (unsigned short*)alloc(3 * WM * 2);
  unsigned short* nmw1T = (unsigned short*)alloc(4 * WM * 2);
  unsigned short* fmw1T = (unsigned short*)alloc(4 * WM * 2);
  unsigned short* fmw2T = (unsigned short*)alloc(4 * WM * 2);
  unsigned short* emw1T = (unsigned short*)alloc(4 * (size_t)256 * HID * 2);

  // ---- prep: convert x; transpose+convert all GEMM weights to bf16 NxK ----
  convert_f32_bf16<<<(unsigned)cdiv(ACT, 256), 256, 0, stream>>>(x, xb, (long)ACT);
  for (int l = 0; l < 3; ++l) {
    wprep<<<(unsigned)cdiv(WM, 256), 256, 0, stream>>>(cw1 + l * WM, cw1T + l * WM, HID, HID);
    wprep<<<(unsigned)cdiv(WM, 256), 256, 0, stream>>>(cw2 + l * WM, cw2T + l * WM, HID, HID);
    wprep<<<(unsigned)cdiv(WM, 256), 256, 0, stream>>>(kw1 + l * WM, kw1T + l * WM, HID, HID);
    wprep<<<(unsigned)cdiv(WM, 256), 256, 0, stream>>>(kw2 + l * WM, kw2T + l * WM, HID, HID);
  }
  for (int k = 0; k < NUM_EXPERTS; ++k) {
    wprep<<<(unsigned)cdiv(WM, 256), 256, 0, stream>>>(nm_w1 + k * WM, nmw1T + k * WM, HID, HID);
    wprep<<<(unsigned)cdiv(WM, 256), 256, 0, stream>>>(fm_w1 + k * WM, fmw1T + k * WM, HID, HID);
    wprep<<<(unsigned)cdiv(WM, 256), 256, 0, stream>>>(fm_w2 + k * WM, fmw2T + k * WM, HID, HID);
    wprep<<<(unsigned)cdiv((size_t)256 * HID, 256), 256, 0, stream>>>(
        em_w1 + (size_t)k * 256 * HID, emw1T + (size_t)k * 256 * HID, 256, HID);
  }

  // ---- base encoder: Z = GIN(x; cw*) ----
  run_encoder(xb, hbA, hbB, zb, t1b, agg, cw1T, cw2T, cb1, cb2, ceps,
              src, dst, nullptr, 0, stream);
  hipMemcpyAsync(Zb, hbA, ACT * 2, hipMemcpyDeviceToDevice, stream);

  // h_orig = mean_pool(Z)
  hipMemsetAsync(pool, 0, (size_t)NUM_GRAPHS * HID * 4, stream);
  hipMemsetAsync(cnt, 0, (size_t)NUM_GRAPHS * 4, stream);
  pool_accum<<<(unsigned)cdiv((long)N_NODES * 32, 256), 256, 0, stream>>>(Zb, batch, pool, cnt);
  pool_final<<<(unsigned)cdiv(NUM_GRAPHS * HID, 256), 256, 0, stream>>>(pool, cnt, out_horig, HID);

  // edge_feat (shared across experts)
  build_edge_feat<<<(unsigned)cdiv((long)N_EDGES * 32, 256), 256, 0, stream>>>(Zb, src, dst, efb);

  for (int k = 0; k < NUM_EXPERTS; ++k) {
    // node mask: HC(relu(Z@nm_w1+b1)@nm_w2 + b2)
    maskhead_wmma<<<N_NODES / 16, 256, 0, stream>>>(
        Zb, nmw1T + k * WM, nm_b1 + k * HID, nm_w2 + k * HID, nm_b2 + k,
        out_nodemask + k, 4, HID, 1000u + k);
    // edge mask: HC(relu(ef@em_w1+b1)@em_w2 + b2), K=256, fully fused
    maskhead_wmma<<<N_EDGES / 16, 256, 0, stream>>>(
        efb, emw1T + (size_t)k * 256 * HID, em_b1 + k * HID, em_w2 + k * HID, em_b2 + k,
        out_edgemask + k, 4, 256, 2000u + k);
    // feat mask: t1 = relu(Z@fm_w1+b1); mask = HC(t1@fm_w2+b2)
    gemm_bf16_wmma<<<N_NODES / 16, 256, 0, stream>>>(
        Zb, fmw1T + k * WM, fm_b1 + k * HID, t1b, HID, nullptr, 0,
        N_NODES, HID, HID, 1, 0, 0u);
    gemm_bf16_wmma<<<N_NODES / 16, 256, 0, stream>>>(
        t1b, fmw2T + k * WM, fm_b2 + k * HID, nullptr, 0,
        out_featmask + k * 128, 512, N_NODES, HID, NUM_FEAT, 0, 1, 3000u + k);
    // masked_x = x * node_mask * feat_mask
    masked_x_kernel<<<(unsigned)cdiv(ACT, 256), 256, 0, stream>>>(
        xb, out_nodemask + k, out_featmask + k * 128, mxb);
    // masked encoder with edge weights
    run_encoder(mxb, hbA, hbB, zb, t1b, agg, kw1T, kw2T, kb1, kb2, keps,
                src, dst, out_edgemask + k, 4, stream);
    // h_stable[:,k,:] = mean_pool(masked_Z)
    hipMemsetAsync(pool, 0, (size_t)NUM_GRAPHS * HID * 4, stream);
    hipMemsetAsync(cnt, 0, (size_t)NUM_GRAPHS * 4, stream);
    pool_accum<<<(unsigned)cdiv((long)N_NODES * 32, 256), 256, 0, stream>>>(hbA, batch, pool, cnt);
    pool_final<<<(unsigned)cdiv(NUM_GRAPHS * HID, 256), 256, 0, stream>>>(
        pool, cnt, out_hstable + k * 128, 4 * 128);
    // expert logits
    logits_kernel<<<NUM_GRAPHS, 32, 0, stream>>>(
        out_hstable + k * 128, 4 * 128,
        cls_w + (size_t)k * HID * NUM_CLASSES, cls_b + k * NUM_CLASSES,
        out_logits + k * NUM_CLASSES, 4 * NUM_CLASSES);
  }
  (void)in_sizes; (void)n_in; (void)out_size; (void)ws_size;
}
